// MixtureOfExperts_38482906972896
// MI455X (gfx1250) — matrix operand order repaired and checked by reading, not compiled, not run on hardware
//
#include <hip/hip_runtime.h>

// ---------------- problem constants ----------------
#define TOKENS   8192            // B*S
#define HDIM     1024
#define IDIM     4096
#define NEXP     8
#define TOPK     2
#define CAP      8192            // per-expert bucket capacity (worst case)

typedef __attribute__((ext_vector_type(4)))  unsigned short ushort4v;
typedef __attribute__((ext_vector_type(8)))  unsigned short ushort8v;
typedef __attribute__((ext_vector_type(16))) unsigned short ushort16v;
typedef __attribute__((ext_vector_type(16))) __bf16        v16bf;
typedef __attribute__((ext_vector_type(8)))  float         v8f;

// async global->LDS copy (ASYNCcnt-tracked), GVS addressing: saddr base + 32-bit voffset
#define ASYNC_B128(ldsoff, voff, sbase)                                        \
  asm volatile("global_load_async_to_lds_b128 %0, %1, %2"                      \
               :: "v"(ldsoff), "v"(voff), "s"(sbase) : "memory")

#if __has_builtin(__builtin_amdgcn_s_wait_asynccnt)
#define WAIT_ASYNCCNT(n) __builtin_amdgcn_s_wait_asynccnt(n)
#else
#define WAIT_ASYNCCNT(n) asm volatile("s_wait_asynccnt %0" :: "n"(n) : "memory")
#endif

__device__ __forceinline__ unsigned short f2bf(float f) {
  unsigned int u = __float_as_uint(f);
  unsigned int r = u + 0x7FFFu + ((u >> 16) & 1u);   // round-to-nearest-even
  return (unsigned short)(r >> 16);
}

__device__ __forceinline__ v8f wmma_bf16(v16bf a, v16bf b, v8f c) {
  return __builtin_amdgcn_wmma_f32_16x16x32_bf16(
      /*neg_a=*/false, a, /*neg_b=*/false, b,
      /*c_mod=*/(short)0, c, /*reuse_a=*/false, /*reuse_b=*/false);
}

// A fragment: lane holds row M=lane&15; VGPR0-3 = K[8*kh..], VGPR4-7 = K[16+8*kh..]
__device__ __forceinline__ v16bf make_a_frag(const unsigned short* As, int row, int kh) {
  ushort8v c0 = *(const ushort8v*)(As + row * 32 + kh * 8);
  ushort8v c1 = *(const ushort8v*)(As + row * 32 + 16 + kh * 8);
  ushort16v u;
#pragma unroll
  for (int i = 0; i < 8; ++i) { u[i] = c0[i]; u[i + 8] = c1[i]; }
  return __builtin_bit_cast(v16bf, u);
}

// B fragment: lane holds col N=n+(lane&15); VGPR0-3 = K[8*kh..], VGPR4-7 = K[16+8*kh..]
__device__ __forceinline__ v16bf make_b_frag(const unsigned short* Bp, int lane, int n) {
  int col = n + (lane & 15), kh = lane >> 4;
  ushort16v u;
#pragma unroll
  for (int i = 0; i < 8; ++i) { u[i] = Bp[(kh * 8 + i) * 128 + col]; u[i + 8] = Bp[(16 + kh * 8 + i) * 128 + col]; }
  return __builtin_bit_cast(v16bf, u);
}

// ---------------- kernel 0: zero per-launch state ----------------
__global__ void moe_zero_kernel(int* __restrict__ counts) {
  if (threadIdx.x < NEXP) counts[threadIdx.x] = 0;
}

// ---------------- kernel 1a: clip + fp32 -> bf16 (activations) ----------------
__global__ __launch_bounds__(256) void moe_convert_x_kernel(
    const float* __restrict__ x, unsigned short* __restrict__ xf) {
  size_t i = ((size_t)blockIdx.x * 256 + threadIdx.x) * 4;
  float4 v = *(const float4*)(x + i);
  v.x = fminf(fmaxf(v.x, -10.f), 10.f); v.y = fminf(fmaxf(v.y, -10.f), 10.f);
  v.z = fminf(fmaxf(v.z, -10.f), 10.f); v.w = fminf(fmaxf(v.w, -10.f), 10.f);
  ushort4v p = {f2bf(v.x), f2bf(v.y), f2bf(v.z), f2bf(v.w)};
  *(ushort4v*)(xf + i) = p;
}

// ---------------- kernel 1b: fp32 -> bf16 (weights, no clip) ----------------
__global__ __launch_bounds__(256) void moe_convert_w_kernel(
    const float* __restrict__ src, unsigned short* __restrict__ dst) {
  size_t i = ((size_t)blockIdx.x * 256 + threadIdx.x) * 4;
  float4 v = *(const float4*)(src + i);
  ushort4v p = {f2bf(v.x), f2bf(v.y), f2bf(v.z), f2bf(v.w)};
  *(ushort4v*)(dst + i) = p;
}

// ---------------- kernel 2: router (one wave32 per token) ----------------
__global__ __launch_bounds__(256) void moe_router_kernel(
    const float* __restrict__ x, const float* __restrict__ rw,
    int* __restrict__ top_i, float* __restrict__ top_w,
    float* __restrict__ lse2, float* __restrict__ probs_out,
    int* __restrict__ counts, int* __restrict__ bucket_tok,
    int* __restrict__ bucket_slot) {
  int t    = blockIdx.x * 8 + (threadIdx.x >> 5);
  int lane = threadIdx.x & 31;
  if (t >= TOKENS) return;

  float p[NEXP];
#pragma unroll
  for (int e = 0; e < NEXP; ++e) p[e] = 0.f;

  const float* xr = x + (size_t)t * HDIM;
  for (int i = 0; i < HDIM / 32; ++i) {
    int h = i * 32 + lane;
    float xv = xr[h];
    xv = fminf(fmaxf(xv, -10.0f), 10.0f);
    const float* rwr = rw + (size_t)h * NEXP;
#pragma unroll
    for (int e = 0; e < NEXP; ++e) p[e] += xv * rwr[e];
  }
#pragma unroll
  for (int e = 0; e < NEXP; ++e)
    for (int off = 16; off > 0; off >>= 1)
      p[e] += __shfl_xor(p[e], off, 32);

  if (lane == 0) {
    float mx = p[0];
#pragma unroll
    for (int e = 1; e < NEXP; ++e) mx = fmaxf(mx, p[e]);
    float pr[NEXP], se = 0.f;
#pragma unroll
    for (int e = 0; e < NEXP; ++e) { pr[e] = __expf(p[e] - mx); se += pr[e]; }
    float lse = mx + __logf(se);
    lse2[t] = lse * lse;
    float inv = 1.0f / se;
#pragma unroll
    for (int e = 0; e < NEXP; ++e) { pr[e] *= inv; probs_out[t * NEXP + e] = pr[e]; }
    int i0 = 0;
#pragma unroll
    for (int e = 1; e < NEXP; ++e) if (pr[e] > pr[i0]) i0 = e;
    int i1 = (i0 == 0) ? 1 : 0;
#pragma unroll
    for (int e = 0; e < NEXP; ++e) if (e != i0 && pr[e] > pr[i1]) i1 = e;
    float w0 = pr[i0], w1 = pr[i1];
    float ws = w0 + w1; w0 /= ws; w1 /= ws;
    top_i[t * 2] = i0; top_i[t * 2 + 1] = i1;
    top_w[t * 2] = w0; top_w[t * 2 + 1] = w1;
    int ei[2] = {i0, i1};
#pragma unroll
    for (int k = 0; k < 2; ++k) {
      int e = ei[k];
      int pos = atomicAdd(&counts[e], 1);
      bucket_tok[e * CAP + pos]  = t;
      bucket_slot[e * CAP + pos] = t * 2 + k;
    }
  }
}

// ---------------- kernel 3: loss reductions + expert base offsets ----------------
__global__ __launch_bounds__(256) void moe_finalize_kernel(
    const float* __restrict__ lse2, const float* __restrict__ probs,
    const int* __restrict__ counts, int* __restrict__ base,
    float* __restrict__ loss_out) {
  __shared__ float red[256];
  __shared__ float mp[NEXP];
  __shared__ float zsh;
  int tid = threadIdx.x;

  float s = 0.f;
  for (int i = tid; i < TOKENS; i += 256) s += lse2[i];
  red[tid] = s; __syncthreads();
  for (int o = 128; o > 0; o >>= 1) {
    if (tid < o) red[tid] += red[tid + o];
    __syncthreads();
  }
  if (tid == 0) zsh = red[0];

  for (int e = 0; e < NEXP; ++e) {
    __syncthreads();
    s = 0.f;
    for (int i = tid; i < TOKENS; i += 256) s += probs[i * NEXP + e];
    red[tid] = s; __syncthreads();
    for (int o = 128; o > 0; o >>= 1) {
      if (tid < o) red[tid] += red[tid + o];
      __syncthreads();
    }
    if (tid == 0) mp[e] = red[0];
  }
  __syncthreads();
  if (tid == 0) {
    int b = 0; float lb = 0.f;
    for (int e = 0; e < NEXP; ++e) {
      base[e] = b; b += counts[e];
      float frac = ((float)counts[e] / (float)TOKENS) * 0.5f;   // /TOPK
      lb += frac * (mp[e] / (float)TOKENS);
    }
    loss_out[0] = 0.001f * (float)NEXP * lb;   // lb_loss
    loss_out[1] = zsh / (float)TOKENS;         // z_loss
  }
}

// ---------------- kernel 4: GEMM1  act = silu(x@Wg)*(x@Wu), 128x128 tile, async DB ----------------
__global__ __launch_bounds__(256) void moe_gemm1_kernel(
    const unsigned short* __restrict__ xf, const unsigned short* __restrict__ wg,
    const unsigned short* __restrict__ wu, const int* __restrict__ counts,
    const int* __restrict__ base, const int* __restrict__ bucket_tok,
    unsigned short* __restrict__ act) {
  int e        = blockIdx.x >> 6;      // 64 row-tiles of 128 per expert
  int row_tile = blockIdx.x & 63;
  int cnt      = counts[e];
  if (row_tile * 128 >= cnt) return;
  int col0 = blockIdx.y * 128;

  __shared__ __align__(32) unsigned short As[2][128 * 32];
  __shared__ __align__(32) unsigned short Bg[2][32 * 128];
  __shared__ __align__(32) unsigned short Bu[2][32 * 128];

  int tid = threadIdx.x;
  // A staging: row = tid>>1, 32B at ushort col (tid&1)*16
  int arow = tid >> 1, acol = (tid & 1) * 16;
  int rg = row_tile * 128 + arow; if (rg >= cnt) rg = cnt - 1;
  int tok = bucket_tok[e * CAP + rg];
  unsigned long long a_base = (unsigned long long)xf;
  unsigned a_off0 = ((unsigned)tok * HDIM + acol) * 2u;      // + k0*2
  // B staging: row = tid>>3, 32B at ushort col (tid&7)*16
  int brow = tid >> 3, bcol = (tid & 7) * 16;
  unsigned long long g_base = (unsigned long long)(wg + (size_t)e * HDIM * IDIM);
  unsigned long long u_base = (unsigned long long)(wu + (size_t)e * HDIM * IDIM);
  unsigned b_off0 = ((unsigned)brow * IDIM + (unsigned)(col0 + bcol)) * 2u;  // + k0*IDIM*2

  auto issue = [&](int buf, int k0) {
    unsigned a_lds = (unsigned)(size_t)&As[buf][arow * 32 + acol];
    unsigned av = a_off0 + (unsigned)k0 * 2u;
    ASYNC_B128(a_lds,      av,      a_base);
    ASYNC_B128(a_lds + 16, av + 16, a_base);
    unsigned bo = b_off0 + (unsigned)k0 * (IDIM * 2u);
    unsigned g_lds = (unsigned)(size_t)&Bg[buf][brow * 128 + bcol];
    ASYNC_B128(g_lds,      bo,      g_base);
    ASYNC_B128(g_lds + 16, bo + 16, g_base);
    unsigned ulds = (unsigned)(size_t)&Bu[buf][brow * 128 + bcol];
    ASYNC_B128(ulds,      bo,      u_base);
    ASYNC_B128(ulds + 16, bo + 16, u_base);
  };

  int lane = tid & 31, w = tid >> 5;
  int wm = (w & 3) * 32;        // wave: 32 rows
  int wn = (w >> 2) * 64;       // wave: 64 cols
  int fr = lane & 15, kh = lane >> 4;

  v8f accg[2][4], accu[2][4];
#pragma unroll
  for (int a = 0; a < 2; ++a)
#pragma unroll
    for (int c = 0; c < 4; ++c) { accg[a][c] = (v8f)0.f; accu[a][c] = (v8f)0.f; }

  issue(0, 0);
  for (int k0 = 0, it = 0; k0 < HDIM; k0 += 32, ++it) {
    int cur = it & 1;
    if (k0 + 32 < HDIM) { issue(cur ^ 1, k0 + 32); WAIT_ASYNCCNT(6); }
    else                { WAIT_ASYNCCNT(0); }
    __syncthreads();

    v16bf af0 = make_a_frag(&As[cur][0], wm + fr, kh);
    v16bf af1 = make_a_frag(&As[cur][0], wm + 16 + fr, kh);
#pragma unroll
    for (int c = 0; c < 4; ++c) {
      v16bf bg = make_b_frag(&Bg[cur][0], lane, wn + c * 16);
      accg[0][c] = wmma_bf16(af0, bg, accg[0][c]);
      accg[1][c] = wmma_bf16(af1, bg, accg[1][c]);
      v16bf bu = make_b_frag(&Bu[cur][0], lane, wn + c * 16);
      accu[0][c] = wmma_bf16(af0, bu, accu[0][c]);
      accu[1][c] = wmma_bf16(af1, bu, accu[1][c]);
    }
    __syncthreads();
  }

  int b = base[e];
#pragma unroll
  for (int a = 0; a < 2; ++a) {
#pragma unroll
    for (int c = 0; c < 4; ++c) {
#pragma unroll
      for (int j = 0; j < 8; ++j) {
        int rr = wm + a * 16 + j + 8 * kh;          // M = vgpr + 8*(lane>>4)
        int gr = row_tile * 128 + rr;
        if (gr < cnt) {
          float g = accg[a][c][j], u = accu[a][c][j];
          float h = g * __builtin_amdgcn_rcpf(1.0f + __expf(-g)) * u;  // silu(g)*u
          int col = col0 + wn + c * 16 + fr;        // N = lane&15
          act[(size_t)(b + gr) * IDIM + col] = f2bf(h);
        }
      }
    }
  }
}

// ---------------- kernel 5: GEMM2  y[slot] = w * (act @ Wd), 128x128 tile, async DB ----------------
__global__ __launch_bounds__(256) void moe_gemm2_kernel(
    const unsigned short* __restrict__ act, const unsigned short* __restrict__ wd,
    const int* __restrict__ counts, const int* __restrict__ base,
    const int* __restrict__ bucket_slot, const float* __restrict__ top_w,
    float* __restrict__ y_slots) {
  int e        = blockIdx.x >> 6;
  int row_tile = blockIdx.x & 63;
  int cnt      = counts[e];
  if (row_tile * 128 >= cnt) return;
  int col0 = blockIdx.y * 128;
  int bse  = base[e];

  __shared__ __align__(32) unsigned short As[2][128 * 32];
  __shared__ __align__(32) unsigned short Bs[2][32 * 128];

  int tid = threadIdx.x;
  int arow = tid >> 1, acol = (tid & 1) * 16;
  int rg = row_tile * 128 + arow; if (rg >= cnt) rg = cnt - 1;
  unsigned long long a_base = (unsigned long long)act;
  unsigned a_off0 = ((unsigned)(bse + rg) * IDIM + acol) * 2u;
  int brow = tid >> 3, bcol = (tid & 7) * 16;
  unsigned long long d_base = (unsigned long long)(wd + (size_t)e * IDIM * HDIM);
  unsigned b_off0 = ((unsigned)brow * HDIM + (unsigned)(col0 + bcol)) * 2u;

  auto issue = [&](int buf, int k0) {
    unsigned a_lds = (unsigned)(size_t)&As[buf][arow * 32 + acol];
    unsigned av = a_off0 + (unsigned)k0 * 2u;
    ASYNC_B128(a_lds,      av,      a_base);
    ASYNC_B128(a_lds + 16, av + 16, a_base);
    unsigned b_lds = (unsigned)(size_t)&Bs[buf][brow * 128 + bcol];
    unsigned bo = b_off0 + (unsigned)k0 * (HDIM * 2u);
    ASYNC_B128(b_lds,      bo,      d_base);
    ASYNC_B128(b_lds + 16, bo + 16, d_base);
  };

  int lane = tid & 31, w = tid >> 5;
  int wm = (w & 3) * 32, wn = (w >> 2) * 64;
  int fr = lane & 15, kh = lane >> 4;

  v8f acc[2][4];
#pragma unroll
  for (int a = 0; a < 2; ++a)
#pragma unroll
    for (int c = 0; c < 4; ++c) acc[a][c] = (v8f)0.f;

  issue(0, 0);
  for (int k0 = 0, it = 0; k0 < IDIM; k0 += 32, ++it) {
    int cur = it & 1;
    if (k0 + 32 < IDIM) { issue(cur ^ 1, k0 + 32); WAIT_ASYNCCNT(4); }
    else                { WAIT_ASYNCCNT(0); }
    __syncthreads();

    v16bf af0 = make_a_frag(&As[cur][0], wm + fr, kh);
    v16bf af1 = make_a_frag(&As[cur][0], wm + 16 + fr, kh);
#pragma unroll
    for (int c = 0; c < 4; ++c) {
      v16bf bf = make_b_frag(&Bs[cur][0], lane, wn + c * 16);
      acc[0][c] = wmma_bf16(af0, bf, acc[0][c]);
      acc[1][c] = wmma_bf16(af1, bf, acc[1][c]);
    }
    __syncthreads();
  }

#pragma unroll
  for (int a = 0; a < 2; ++a) {
#pragma unroll
    for (int c = 0; c < 4; ++c) {
#pragma unroll
      for (int j = 0; j < 8; ++j) {
        int rr = wm + a * 16 + j + 8 * kh;
        int gr = row_tile * 128 + rr;
        if (gr < cnt) {
          int slot = bucket_slot[e * CAP + gr];
          float wv = top_w[slot];
          int col = col0 + wn + c * 16 + fr;
          y_slots[(size_t)slot * HDIM + col] = wv * acc[a][c][j];
        }
      }
    }
  }
}

// ---------------- kernel 6: combine two slots + clip ----------------
__global__ __launch_bounds__(256) void moe_combine_kernel(
    const float* __restrict__ y_slots, float* __restrict__ out) {
  int i = blockIdx.x * 256 + threadIdx.x;
  int t = i >> 10, h = i & 1023;
  float v = y_slots[(size_t)(t * 2) * HDIM + h] + y_slots[(size_t)(t * 2 + 1) * HDIM + h];
  v = fminf(fmaxf(v, -10.0f), 10.0f);
  out[i] = v;
}

// ---------------- host launcher ----------------
extern "C" void kernel_launch(void* const* d_in, const int* in_sizes, int n_in,
                              void* d_out, int out_size, void* d_ws, size_t ws_size,
                              hipStream_t stream) {
  const float* x       = (const float*)d_in[0];   // (B,S,H) fp32
  const float* rw      = (const float*)d_in[1];   // (H,E)
  const float* w_gate  = (const float*)d_in[2];   // (E,H,I)
  const float* w_up    = (const float*)d_in[3];   // (E,H,I)
  const float* w_down  = (const float*)d_in[4];   // (E,I,H)
  float* out      = (float*)d_out;                          // T*H
  float* loss_out = (float*)d_out + (size_t)TOKENS * HDIM;  // [lb, z]

  const size_t WSZ = (size_t)NEXP * HDIM * IDIM;  // elems per weight tensor

  // ---- workspace layout ----
  char* ws = (char*)d_ws;
  size_t off = 0;
  unsigned short* xf    = (unsigned short*)(ws + off); off += (size_t)TOKENS * HDIM * 2;        // 16 MB
  unsigned short* wg_bf = (unsigned short*)(ws + off); off += WSZ * 2;                          // 67 MB
  unsigned short* wu_bf = (unsigned short*)(ws + off); off += WSZ * 2;                          // 67 MB
  unsigned short* wd_bf = (unsigned short*)(ws + off); off += WSZ * 2;                          // 67 MB
  unsigned short* act   = (unsigned short*)(ws + off); off += (size_t)TOKENS * TOPK * IDIM * 2; // 134 MB
  float* y_slots = (float*)(ws + off); off += (size_t)TOKENS * TOPK * HDIM * 4;                 // 67 MB
  int*   top_i  = (int*)(ws + off);   off += (size_t)TOKENS * 2 * 4;
  float* top_w  = (float*)(ws + off); off += (size_t)TOKENS * 2 * 4;
  float* lse2   = (float*)(ws + off); off += (size_t)TOKENS * 4;
  float* probs  = (float*)(ws + off); off += (size_t)TOKENS * NEXP * 4;
  int* counts      = (int*)(ws + off); off += 256;
  int* base        = (int*)(ws + off); off += 256;
  int* bucket_tok  = (int*)(ws + off); off += (size_t)NEXP * CAP * 4;
  int* bucket_slot = (int*)(ws + off); off += (size_t)NEXP * CAP * 4;
  (void)off; (void)in_sizes; (void)n_in; (void)out_size; (void)ws_size; (void)top_i;

  moe_zero_kernel<<<1, 32, 0, stream>>>(counts);
  moe_convert_x_kernel<<<(TOKENS * HDIM / 4) / 256, 256, 0, stream>>>(x, xf);
  moe_convert_w_kernel<<<(int)((WSZ / 4) / 256), 256, 0, stream>>>(w_gate, wg_bf);
  moe_convert_w_kernel<<<(int)((WSZ / 4) / 256), 256, 0, stream>>>(w_up,   wu_bf);
  moe_convert_w_kernel<<<(int)((WSZ / 4) / 256), 256, 0, stream>>>(w_down, wd_bf);

  moe_router_kernel<<<TOKENS / 8, 256, 0, stream>>>(
      x, rw, top_i, top_w, lse2, probs, counts, bucket_tok, bucket_slot);
  moe_finalize_kernel<<<1, 256, 0, stream>>>(lse2, probs, counts, base, loss_out);

  dim3 g1(NEXP * 64, IDIM / 128);   // 512 x 32, early-exit on count
  moe_gemm1_kernel<<<g1, 256, 0, stream>>>(xf, wg_bf, wu_bf, counts, base, bucket_tok, act);

  dim3 g2(NEXP * 64, HDIM / 128);   // 512 x 8
  moe_gemm2_kernel<<<g2, 256, 0, stream>>>(act, wd_bf, counts, base, bucket_slot, top_w, y_slots);

  moe_combine_kernel<<<(TOKENS * HDIM) / 256, 256, 0, stream>>>(y_slots, out);
}
